// DebertaV2DisentangledSelfAttention_69140383531549
// MI455X (gfx1250) — matrix operand order repaired and checked by reading, not compiled, not run on hardware
//
#include <hip/hip_runtime.h>
#include <hip/hip_bf16.h>

typedef __attribute__((ext_vector_type(16))) _Float16 v16h;
typedef __attribute__((ext_vector_type(8)))  float    v8f;
typedef __attribute__((ext_vector_type(4)))  int      v4i;

#define NHEADS 12
#define DHEAD  64
#define BATCH  2
#define SEQ    2048
#define HID    768
#define KREL   512
#define KV_STRIDE 72   // 64 + 8 f16 pad: breaks 64-bank conflicts for B-operand ds loads

// CDNA5 async global->LDS path (ASYNCcnt), if this clang declares it.
#if defined(__has_builtin)
#if __has_builtin(__builtin_amdgcn_global_load_async_to_lds_b128) && \
    __has_builtin(__builtin_amdgcn_s_wait_asynccnt)
#define HAVE_ASYNC_LDS 1
#endif
#endif
#ifndef HAVE_ASYNC_LDS
#define HAVE_ASYNC_LDS 0
#endif

#if HAVE_ASYNC_LDS
typedef __attribute__((address_space(1))) v4i* as1_v4i_ptr;  // global (printed __device__)
typedef __attribute__((address_space(3))) v4i* as3_v4i_ptr;  // LDS    (printed __shared__)
#endif

// ---------------------------------------------------------------------------
// CDNA5 wave32 WMMA operand loaders (16x16x32 f16 shapes, contiguous k = 16*(lane/16)+e in both A and B)
// ---------------------------------------------------------------------------

// A matrix 16x32 (MxK) f16 from row-major [M x K] source.
// lane<16: M=lane, K=0..15 ; lane>=16: M=lane-16, K=16..31
__device__ __forceinline__ v16h load_a16x32(const _Float16* __restrict__ A, int lda,
                                            int row0, int k0, int lane) {
  const int m  = lane & 15;
  const int kb = (lane >> 4) << 4;  // 0 or 16
  const _Float16* p = A + (size_t)(row0 + m) * lda + k0;
  v16h a;
#pragma unroll
  for (int e = 0; e < 16; ++e) a[e] = p[kb + e];
  return a;
}

// B matrix 32x16 (KxN) sourced from row-major [N x K] (i.e. B = M^T, like nn.Linear W).
// lane<16: N=lane, K=0..15 ; lane>=16: N=lane-16, K=16..31
__device__ __forceinline__ v16h load_b_from_nk(const _Float16* __restrict__ Wt, int ldw,
                                               int col0, int k0, int lane) {
  const int n  = lane & 15;
  const int kb = (lane >> 4) << 4;  // 0 or 16
  const _Float16* p = Wt + (size_t)(col0 + n) * ldw + k0 + kb;
  v16h b;
#pragma unroll
  for (int e = 0; e < 16; ++e) b[e] = p[e];
  return b;
}

// B matrix 32x16 (KxN) sourced from row-major [K x N] (e.g. V for P@V).
__device__ __forceinline__ v16h load_b_from_kn(const _Float16* __restrict__ V, int ldv,
                                               int k0, int col0, int lane) {
  const int n  = lane & 15;
  const int kb = (lane >> 4) << 4;
  const _Float16* p = V + (size_t)(k0 + kb) * ldv + col0 + n;
  v16h b;
#pragma unroll
  for (int e = 0; e < 16; ++e) b[e] = p[(size_t)e * ldv];
  return b;
}

// ---------------------------------------------------------------------------
// Cooperative staging of a 32x64 f16 tile into LDS (row stride KV_STRIDE).
// 128 threads x 2 chunks x 16B = 4 KB tile. Async (ASYNCcnt) when available.
// ---------------------------------------------------------------------------
__device__ __forceinline__ void stage_kv_tile(const _Float16* __restrict__ g,
                                              _Float16* l, int tid) {
#pragma unroll
  for (int c = 0; c < 2; ++c) {
    const int chunk = tid * 2 + c;        // 0..255
    const int row = chunk >> 3;           // 0..31
    const int col = (chunk & 7) << 3;     // 0..56 step 8
    const _Float16* gp = g + (size_t)row * DHEAD + col;
    _Float16* lp = l + row * KV_STRIDE + col;
#if HAVE_ASYNC_LDS
    __builtin_amdgcn_global_load_async_to_lds_b128(
        (as1_v4i_ptr)gp, (as3_v4i_ptr)lp, 0, 0);
#else
    *reinterpret_cast<uint4*>(lp) = *reinterpret_cast<const uint4*>(gp);
#endif
  }
}

// ---------------------------------------------------------------------------
// f32 -> f16 conversion
// ---------------------------------------------------------------------------
__global__ void f32_to_f16_kernel(const float* __restrict__ in, _Float16* __restrict__ out, int n) {
  int i = blockIdx.x * 256 + threadIdx.x;
  if (i < n) out[i] = (_Float16)in[i];
}

// ---------------------------------------------------------------------------
// Projection GEMM:  out[row, col] = sum_k A[row,k] * W[col,k] + bias[col]
// mode 0: row=(b*SEQ+s), store head-major ((b*NH+h)*SEQ+s)*DH+d  (q/k/v)
// mode 1: row=r,        store (h*KREL+r)*DH+d                    (pos tables)
// block = 128 threads (4 waves); each wave -> 16x64 output tile
// ---------------------------------------------------------------------------
__global__ void __launch_bounds__(128) proj_gemm_kernel(
    const _Float16* __restrict__ A, const _Float16* __restrict__ W,
    const float* __restrict__ bias, _Float16* __restrict__ out, int mode) {
  const int lane = threadIdx.x & 31;
  const int wave = threadIdx.x >> 5;
  const int row0 = blockIdx.x * 64 + wave * 16;
  const int col0 = blockIdx.y * 64;

  v8f acc[4] = {};
  for (int k0 = 0; k0 < HID; k0 += 32) {
    v16h a = load_a16x32(A, HID, row0, k0, lane);
#pragma unroll
    for (int t = 0; t < 4; ++t) {
      v16h b = load_b_from_nk(W, HID, col0 + 16 * t, k0, lane);
      acc[t] = __builtin_amdgcn_wmma_f32_16x16x32_f16(false, a, false, b,
                                                      (short)0, acc[t], false, false);
    }
  }

  const int rhalf = (lane >> 4) << 3;
  const int ncol  = lane & 15;
#pragma unroll
  for (int t = 0; t < 4; ++t) {
    const int col = col0 + 16 * t + ncol;
    const int h = col >> 6;
    const int d = col & 63;
    const float bv = bias[col];
#pragma unroll
    for (int r = 0; r < 8; ++r) {
      const int row = row0 + rhalf + r;
      const float val = acc[t][r] + bv;
      size_t idx;
      if (mode == 0) {
        const int bb = row >> 11;          // row / SEQ
        const int s  = row & (SEQ - 1);
        idx = (((size_t)(bb * NHEADS + h)) * SEQ + s) * DHEAD + d;
      } else {
        idx = (((size_t)h) * KREL + row) * DHEAD + d;
      }
      out[idx] = (_Float16)val;
    }
  }
}

// ---------------------------------------------------------------------------
// Relative-bias GEMM: C[bh][s][r] = sum_d Qh[bh][s][d] * Pos[h][r][d], f16 out
// grid: (SEQ/64, KREL/64, B*NH), block 128
// ---------------------------------------------------------------------------
__global__ void __launch_bounds__(128) rel_gemm_kernel(
    const _Float16* __restrict__ Qh, const _Float16* __restrict__ Pos,
    _Float16* __restrict__ Cout) {
  const int lane = threadIdx.x & 31;
  const int wave = threadIdx.x >> 5;
  const int bh = blockIdx.z;
  const int h  = bh % NHEADS;
  const int row0 = blockIdx.x * 64 + wave * 16;
  const int col0 = blockIdx.y * 64;

  const _Float16* A = Qh + (size_t)bh * SEQ * DHEAD;
  const _Float16* P = Pos + (size_t)h * KREL * DHEAD;
  _Float16* C = Cout + (size_t)bh * SEQ * KREL;

  v8f acc[4] = {};
#pragma unroll
  for (int k0 = 0; k0 < DHEAD; k0 += 32) {
    v16h a = load_a16x32(A, DHEAD, row0, k0, lane);
#pragma unroll
    for (int t = 0; t < 4; ++t) {
      v16h b = load_b_from_nk(P, DHEAD, col0 + 16 * t, k0, lane);
      acc[t] = __builtin_amdgcn_wmma_f32_16x16x32_f16(false, a, false, b,
                                                      (short)0, acc[t], false, false);
    }
  }

  const int rhalf = (lane >> 4) << 3;
  const int ncol  = lane & 15;
#pragma unroll
  for (int t = 0; t < 4; ++t)
#pragma unroll
    for (int r = 0; r < 8; ++r)
      C[(size_t)(row0 + rhalf + r) * KREL + col0 + 16 * t + ncol] = (_Float16)acc[t][r];
}

// ---------------------------------------------------------------------------
// Fused flash attention + gathered disentangled bias.
// grid: (SEQ/64, B*NH), block 128 (4 waves). Each wave owns 16 query rows.
// K/V tiles cooperatively staged into LDS (double-buffered, async when avail):
// every tile fetched once per block instead of once per wave.
// ---------------------------------------------------------------------------
__global__ void __launch_bounds__(128) attn_kernel(
    const _Float16* __restrict__ q_ws, const _Float16* __restrict__ k_ws,
    const _Float16* __restrict__ v_ws, const _Float16* __restrict__ c2p,
    const _Float16* __restrict__ p2c, const int* __restrict__ rel_pos,
    const float* __restrict__ mask, float* __restrict__ out) {
  const int tid  = threadIdx.x;
  const int lane = tid & 31;
  const int wave = tid >> 5;
  const int bh = blockIdx.y;
  const int b = bh / NHEADS;
  const int h = bh % NHEADS;
  const int i0 = blockIdx.x * 64 + wave * 16;

  const _Float16* Q   = q_ws + (size_t)bh * SEQ * DHEAD;
  const _Float16* Km  = k_ws + (size_t)bh * SEQ * DHEAD;
  const _Float16* V   = v_ws + (size_t)bh * SEQ * DHEAD;
  const _Float16* C2P = c2p + (size_t)bh * SEQ * KREL;
  const _Float16* P2C = p2c + (size_t)bh * SEQ * KREL;
  const float* msk = mask + (size_t)b * SEQ;

  __shared__ alignas(16) _Float16 kt_lds[2][32 * KV_STRIDE];
  __shared__ alignas(16) _Float16 vt_lds[2][32 * KV_STRIDE];
  __shared__ alignas(16) _Float16 p_stage[4][16 * 32];   // per-wave P (C->A relayout)
  _Float16* pl = p_stage[wave];

  const int rhalf = (lane >> 4) << 3;
  const int ncol  = lane & 15;

  // Q A-operands for the whole i-tile (dh=64 -> two K steps), kept in VGPRs.
  v16h aq0 = load_a16x32(Q, DHEAD, i0, 0, lane);
  v16h aq1 = load_a16x32(Q, DHEAD, i0, 32, lane);

  float mstate[8], lstate[8];
#pragma unroll
  for (int r = 0; r < 8; ++r) { mstate[r] = -3.0e38f; lstate[r] = 0.f; }
  v8f o[4] = {};

  const float inv_scale = 0.07216878364870322f;  // 1/sqrt(3*64)

  // Prime buffer 0 with chunk 0.
  stage_kv_tile(Km, kt_lds[0], tid);
  stage_kv_tile(V,  vt_lds[0], tid);

  const int NCHUNK = SEQ / 32;
  for (int c = 0; c < NCHUNK; ++c) {
    const int j0 = c * 32;
    const int buf = c & 1;
    const bool more = (c + 1) < NCHUNK;

    // Issue next chunk into the other buffer, then wait for the current one.
    if (more) {
      stage_kv_tile(Km + (size_t)(j0 + 32) * DHEAD, kt_lds[buf ^ 1], tid);
      stage_kv_tile(V  + (size_t)(j0 + 32) * DHEAD, vt_lds[buf ^ 1], tid);
    }
#if HAVE_ASYNC_LDS
    // Async loads complete in order: <=4 pending => current buffer's 4 are done.
    if (more) __builtin_amdgcn_s_wait_asynccnt(4);
    else      __builtin_amdgcn_s_wait_asynccnt(0);
#endif
    __syncthreads();

    const _Float16* kl = kt_lds[buf];
    const _Float16* vl = vt_lds[buf];

    // ---- QK^T: two 16x16 score tiles over dh=64, K operand from LDS ----
    v8f s0 = {}, s1 = {};
    {
      v16h b0 = load_b_from_nk(kl, KV_STRIDE, 0, 0, lane);
      s0 = __builtin_amdgcn_wmma_f32_16x16x32_f16(false, aq0, false, b0, (short)0, s0, false, false);
      v16h b1 = load_b_from_nk(kl, KV_STRIDE, 0, 32, lane);
      s0 = __builtin_amdgcn_wmma_f32_16x16x32_f16(false, aq1, false, b1, (short)0, s0, false, false);
      v16h b2 = load_b_from_nk(kl, KV_STRIDE, 16, 0, lane);
      s1 = __builtin_amdgcn_wmma_f32_16x16x32_f16(false, aq0, false, b2, (short)0, s1, false, false);
      v16h b3 = load_b_from_nk(kl, KV_STRIDE, 16, 32, lane);
      s1 = __builtin_amdgcn_wmma_f32_16x16x32_f16(false, aq1, false, b3, (short)0, s1, false, false);
    }

    // ---- gathered c2p/p2c bias + scale + mask + clip (L2-resident gathers) ----
#pragma unroll
    for (int t = 0; t < 2; ++t) {
      const int j = j0 + 16 * t + ncol;
      const float mv = msk[j];
#pragma unroll
      for (int r = 0; r < 8; ++r) {
        const int i = i0 + rhalf + r;
        int rij = rel_pos[(size_t)i * SEQ + j];
        int rji = rel_pos[(size_t)j * SEQ + i];
        rij = rij < 0 ? 0 : (rij > KREL - 1 ? KREL - 1 : rij);
        rji = rji < 0 ? 0 : (rji > KREL - 1 ? KREL - 1 : rji);
        float sc = (t == 0 ? s0[r] : s1[r]);
        sc += (float)C2P[(size_t)i * KREL + rij];
        sc += (float)P2C[(size_t)j * KREL + rji];
        sc = sc * inv_scale + mv;
        sc = fminf(fmaxf(sc, -50.f), 50.f);
        if (t == 0) s0[r] = sc; else s1[r] = sc;
      }
    }

    // ---- online softmax (rows live in 16-lane halves of the C layout) ----
    float corr[8];
#pragma unroll
    for (int r = 0; r < 8; ++r) {
      float mx = fmaxf(s0[r], s1[r]);
#pragma unroll
      for (int off = 8; off >= 1; off >>= 1) mx = fmaxf(mx, __shfl_xor(mx, off, 16));
      const float mnew = fmaxf(mstate[r], mx);
      corr[r] = __expf(mstate[r] - mnew);
      mstate[r] = mnew;
      s0[r] = __expf(s0[r] - mnew);
      s1[r] = __expf(s1[r] - mnew);
      float sm = s0[r] + s1[r];
#pragma unroll
      for (int off = 8; off >= 1; off >>= 1) sm += __shfl_xor(sm, off, 16);
      lstate[r] = lstate[r] * corr[r] + sm;
    }
#pragma unroll
    for (int t = 0; t < 4; ++t)
#pragma unroll
      for (int r = 0; r < 8; ++r) o[t][r] *= corr[r];

    // ---- stage P to LDS as f16 row-major 16x32, reload in A layout ----
#pragma unroll
    for (int r = 0; r < 8; ++r) {
      pl[(rhalf + r) * 32 + ncol]      = (_Float16)s0[r];
      pl[(rhalf + r) * 32 + 16 + ncol] = (_Float16)s1[r];
    }
    v16h ap = load_a16x32(pl, 32, 0, 0, lane);

    // ---- P @ V accumulation over dh (4 x 16-wide N tiles), V from LDS ----
#pragma unroll
    for (int t = 0; t < 4; ++t) {
      v16h bv = load_b_from_kn(vl, KV_STRIDE, 0, 16 * t, lane);
      o[t] = __builtin_amdgcn_wmma_f32_16x16x32_f16(false, ap, false, bv, (short)0, o[t], false, false);
    }

    __syncthreads();  // all waves done reading buf before it is re-staged
  }

  // ---- normalize + store ctx: out[b][i][h*64+d] ----
#pragma unroll
  for (int r = 0; r < 8; ++r) {
    const int i = i0 + rhalf + r;
    const float inv_l = 1.0f / lstate[r];
#pragma unroll
    for (int t = 0; t < 4; ++t)
      out[((size_t)b * SEQ + i) * HID + h * DHEAD + 16 * t + ncol] = o[t][r] * inv_l;
  }
}

// ---------------------------------------------------------------------------
extern "C" void kernel_launch(void* const* d_in, const int* in_sizes, int n_in,
                              void* d_out, int out_size, void* d_ws, size_t ws_size,
                              hipStream_t stream) {
  (void)in_sizes; (void)n_in; (void)out_size; (void)ws_size;

  const float* hidden = (const float*)d_in[0];
  const float* amask  = (const float*)d_in[1];
  const float* rel    = (const float*)d_in[2];
  const float* Wq = (const float*)d_in[3];
  const float* bq = (const float*)d_in[4];
  const float* Wk = (const float*)d_in[5];
  const float* bk = (const float*)d_in[6];
  const float* Wv = (const float*)d_in[7];
  const float* bv = (const float*)d_in[8];
  const int* rel_pos = (const int*)d_in[9];
  float* out = (float*)d_out;

  char* ws = (char*)d_ws;
  size_t off = 0;
  auto alloc = [&](size_t bytes) -> void* {
    void* p = (void*)(ws + off);
    off += (bytes + 255) & ~(size_t)255;
    return p;
  };

  _Float16* h16   = (_Float16*)alloc((size_t)BATCH * SEQ * HID * 2);
  _Float16* wq16  = (_Float16*)alloc((size_t)HID * HID * 2);
  _Float16* wk16  = (_Float16*)alloc((size_t)HID * HID * 2);
  _Float16* wv16  = (_Float16*)alloc((size_t)HID * HID * 2);
  _Float16* rel16 = (_Float16*)alloc((size_t)KREL * HID * 2);
  _Float16* q16   = (_Float16*)alloc((size_t)BATCH * NHEADS * SEQ * DHEAD * 2);
  _Float16* k16   = (_Float16*)alloc((size_t)BATCH * NHEADS * SEQ * DHEAD * 2);
  _Float16* v16   = (_Float16*)alloc((size_t)BATCH * NHEADS * SEQ * DHEAD * 2);
  _Float16* pk16  = (_Float16*)alloc((size_t)NHEADS * KREL * DHEAD * 2);
  _Float16* pq16  = (_Float16*)alloc((size_t)NHEADS * KREL * DHEAD * 2);
  _Float16* c2p   = (_Float16*)alloc((size_t)BATCH * NHEADS * SEQ * KREL * 2);
  _Float16* p2c   = (_Float16*)alloc((size_t)BATCH * NHEADS * SEQ * KREL * 2);

  // ---- f32 -> f16 conversions ----
  auto cvt = [&](const float* src, _Float16* dst, int n) {
    f32_to_f16_kernel<<<dim3((n + 255) / 256), dim3(256), 0, stream>>>(src, dst, n);
  };
  cvt(hidden, h16, BATCH * SEQ * HID);
  cvt(Wq, wq16, HID * HID);
  cvt(Wk, wk16, HID * HID);
  cvt(Wv, wv16, HID * HID);
  cvt(rel, rel16, KREL * HID);

  const dim3 blk(128);

  // ---- q/k/v projections (head-major f16) ----
  proj_gemm_kernel<<<dim3(BATCH * SEQ / 64, HID / 64), blk, 0, stream>>>(h16, wq16, bq, q16, 0);
  proj_gemm_kernel<<<dim3(BATCH * SEQ / 64, HID / 64), blk, 0, stream>>>(h16, wk16, bk, k16, 0);
  proj_gemm_kernel<<<dim3(BATCH * SEQ / 64, HID / 64), blk, 0, stream>>>(h16, wv16, bv, v16, 0);

  // ---- positional projections: pos_key = rel @ Wk^T + bk ; pos_query = rel @ Wq^T + bq ----
  proj_gemm_kernel<<<dim3(KREL / 64, HID / 64), blk, 0, stream>>>(rel16, wk16, bk, pk16, 1);
  proj_gemm_kernel<<<dim3(KREL / 64, HID / 64), blk, 0, stream>>>(rel16, wq16, bq, pq16, 1);

  // ---- c2p_full = q . pos_key^T ; p2c_full = k . pos_query^T  (f16, L2-resident) ----
  rel_gemm_kernel<<<dim3(SEQ / 64, KREL / 64, BATCH * NHEADS), blk, 0, stream>>>(q16, pk16, c2p);
  rel_gemm_kernel<<<dim3(SEQ / 64, KREL / 64, BATCH * NHEADS), blk, 0, stream>>>(k16, pq16, p2c);

  // ---- fused attention ----
  attn_kernel<<<dim3(SEQ / 64, BATCH * NHEADS), blk, 0, stream>>>(
      q16, k16, v16, c2p, p2c, rel_pos, amask, out);
}
